// ModulatedDeformConvPack_14516989461141
// MI455X (gfx1250) — matrix operand order repaired and checked by reading, not compiled, not run on hardware
//
#include <hip/hip_runtime.h>
#include <hip/hip_bf16.h>
#include <math.h>

// ---------------------------------------------------------------------------
// ModulatedDeformConv for MI455X / gfx1250 (wave32, WMMA)
//   B=4, C=64, O=64, H=W=128, K=3, S=1, P=1, D=1
// Pipeline:
//   pack_kernel        : swizzle weight -> bf16 A-fragments, w_om -> f32 A-frags
//   offset_conv_kernel : om = conv3x3(x, w_om)+b_om  via v_wmma_f32_16x16x4_f32
//                        (tap-structured loop, clamp+mask loads, no EXEC toggling)
//   deform_gemm_kernel : bilinear sample (fp32) -> bf16 LDS tap tiles
//                        (double-buffered, XOR bank swizzle) ->
//                        v_wmma_f32_16x16x32_bf16 over K=C*9=576
// ---------------------------------------------------------------------------

typedef __attribute__((ext_vector_type(16))) __bf16 v16bf;
typedef __attribute__((ext_vector_type(8)))  float  v8f;
typedef __attribute__((ext_vector_type(2)))  float  v2f;
typedef __attribute__((ext_vector_type(4)))  unsigned int u32x4;

#define BB 4
#define CC 64
#define OO 64
#define HH 128
#define WW 128
#define HWSZ (HH * WW)          // 16384
#define KK2 9
#define CK (CC * KK2)           // 576
#define NCHUNK (CK / 32)        // 18  (bf16 WMMA K-chunks)
#define NSTEP4 (CK / 4)         // 144 (f32 WMMA K-steps)

// ---- workspace layout (bytes) ----
#define WS_WA    0                                   // 4*18*32*16 bf16 = 73728 B
#define WS_WOM   73728                               // 2*144*32*2 f32  = 73728 B
#define WS_DY    147456                              // 4*9*16384 f32   = 2359296 B
#define WS_DX    (WS_DY + 2359296)
#define WS_MASK  (WS_DX + 2359296)

// round-to-nearest-even float -> bf16 (as raw u16), avoids __bf16 scalar math
__device__ __forceinline__ unsigned short f2bf(float f) {
    unsigned int u = __float_as_uint(f);
    u += 0x7FFFu + ((u >> 16) & 1u);
    return (unsigned short)(u >> 16);
}

// ---------------------------------------------------------------------------
// Kernel 0: pack weights into WMMA A-fragment order.
//  bf16 A (16x32): lane<16 -> M=lane, K slots {0..7,16..23};
//                  lane>=16 -> M=lane-16, K slots {8..15,24..31}
//  K index ordering: ck = k*64 + c   (k = kernel tap, c = input channel)
// ---------------------------------------------------------------------------
__global__ __launch_bounds__(256) void pack_kernel(
    const float* __restrict__ weight,   // (O, C, 3, 3)
    const float* __restrict__ w_om,     // (27, C, 3, 3)
    unsigned short* __restrict__ wA,    // [4][18][32][16] bf16
    float* __restrict__ wOm)            // [2][144][32][2] f32
{
    int e = blockIdx.x * 256 + threadIdx.x;

    if (e < 4 * NCHUNK * 32 * 16) {
        int s  = e & 15;
        int l  = (e >> 4) & 31;
        int kc = (e >> 9) % NCHUNK;
        int m  = e / (16 * 32 * NCHUNK);
        int M  = l & 15;
        int Kl = (l < 16) ? ((s < 8) ? s : 16 + (s - 8))
                          : ((s < 8) ? 8 + s : 24 + (s - 8));
        int ck = kc * 32 + Kl;
        int k  = ck >> 6;          // tap 0..8
        int c  = ck & 63;          // channel
        int o  = m * 16 + M;
        wA[e] = f2bf(weight[o * CK + c * KK2 + k]);
    }

    if (e < 2 * NSTEP4 * 32 * 2) {
        int j  = e & 1;
        int l  = (e >> 1) & 31;
        int st = (e >> 6) % NSTEP4;
        int m  = e / (2 * 32 * NSTEP4);
        int M  = l & 15;
        int koff = (l < 16) ? j : 2 + j;     // f32 A 16x4: lane<16 K={0,1}, lane>=16 K={2,3}
        int ck = st * 4 + koff;
        int k  = ck >> 6;
        int c  = ck & 63;
        int row = m * 16 + M;
        wOm[e] = (row < 27) ? w_om[row * CK + c * KK2 + k] : 0.0f;
    }
}

// ---------------------------------------------------------------------------
// Kernel 1: offset conv (M=27 -> 2x16 tiles, K=576, N = pixels) in fp32 WMMA.
// One block = one (b, h) row (128 pixels); 8 waves, each owns 16 pixels.
// Tap-structured K loop: ky/kx and the pad test are per-tap constants; inner
// 16 steps just stride pointers. Pad handled by clamped address * {0,1} scale
// so loads are unconditional (EXEC stays all-ones around the WMMAs).
// ---------------------------------------------------------------------------
__global__ __launch_bounds__(256) void offset_conv_kernel(
    const float* __restrict__ x,
    const float* __restrict__ wOm,       // packed [2][144][32][2]
    const float* __restrict__ w_om_raw,  // fallback path only
    const float* __restrict__ b_om,
    float* __restrict__ dyw, float* __restrict__ dxw, float* __restrict__ mw)
{
    const int b    = blockIdx.x >> 7;
    const int h    = blockIdx.x & 127;
    const int wave = threadIdx.x >> 5;
    const int lane = threadIdx.x & 31;
    const int w0   = wave * 16;
    const int n    = lane & 15;
    const int hi   = lane >> 4;
    const float* xb = x + b * CC * HWSZ;

    float a0[8], a1[8];

#if __has_builtin(__builtin_amdgcn_wmma_f32_16x16x4_f32)
    v8f acc0 = {}; v8f acc1 = {};
#pragma unroll
    for (int kk = 0; kk < KK2; ++kk) {
        const int ky = kk / 3, kx = kk % 3;
        const int row = h + ky - 1;
        const int col = w0 + n + kx - 1;
        const float sc = ((unsigned)row < HH && (unsigned)col < WW) ? 1.0f : 0.0f;
        const int rowc = min(max(row, 0), HH - 1);
        const int colc = min(max(col, 0), WW - 1);
        const float* p  = xb + (hi * 2) * HWSZ + rowc * WW + colc;
        const float* aA = wOm + (kk * 16 * 32 + lane) * 2;
        const float* aB = aA + NSTEP4 * 32 * 2;
#pragma unroll 4
        for (int cs = 0; cs < 16; ++cs) {
            v2f bfrag = { p[0] * sc, p[HWSZ] * sc };
            v2f af0 = *(const v2f*)(aA);
            v2f af1 = *(const v2f*)(aB);
            acc0 = __builtin_amdgcn_wmma_f32_16x16x4_f32(false, af0, false, bfrag,
                                                         (short)0, acc0, false, false);
            acc1 = __builtin_amdgcn_wmma_f32_16x16x4_f32(false, af1, false, bfrag,
                                                         (short)0, acc1, false, false);
            p  += 4 * HWSZ;
            aA += 64;
            aB += 64;
        }
    }
    for (int r = 0; r < 8; ++r) { a0[r] = acc0[r]; a1[r] = acc1[r]; }
#else
    // scalar fallback mimicking the WMMA D-fragment mapping
    for (int r = 0; r < 8; ++r) { a0[r] = 0.0f; a1[r] = 0.0f; }
    {
        const int pix = w0 + n;
        for (int ck = 0; ck < CK; ++ck) {
            int kk = ck >> 6, c = ck & 63;
            int ky = kk / 3, kx = kk % 3;
            int row = h + ky - 1, col = pix + kx - 1;
            float xv = ((unsigned)row < HH && (unsigned)col < WW)
                           ? xb[c * HWSZ + row * WW + col] : 0.0f;
            for (int r = 0; r < 8; ++r) {
                int ch0 = r + hi * 8;
                int ch1 = ch0 + 16;
                a0[r] += xv * w_om_raw[ch0 * CK + c * KK2 + kk];
                if (ch1 < 27) a1[r] += xv * w_om_raw[ch1 * CK + c * KK2 + kk];
            }
        }
    }
#endif

    // Epilogue: route om channels. ch = 2k -> dy, 2k+1 -> dx, 18+k -> sigmoid mask
    const int pix  = w0 + n;
    const int base = h * WW + pix;
    for (int r = 0; r < 8; ++r) {
        int ch0 = r + hi * 8;            // 0..15
        int ch1 = ch0 + 16;              // 16..31 (valid < 27)
        {
            float v = a0[r] + b_om[ch0];
            int k = ch0 >> 1;
            float* dst = (ch0 & 1) ? dxw : dyw;
            dst[(b * KK2 + k) * HWSZ + base] = v;
        }
        if (ch1 < 27) {
            float v = a1[r] + b_om[ch1];
            if (ch1 < 18) {
                int k = ch1 >> 1;
                float* dst = (ch1 & 1) ? dxw : dyw;
                dst[(b * KK2 + k) * HWSZ + base] = v;
            } else {
                mw[(b * KK2 + (ch1 - 18)) * HWSZ + base] =
                    1.0f / (1.0f + __expf(-v));
            }
        }
    }
}

// ---------------------------------------------------------------------------
// Kernel 2: fused bilinear sampling + modulated GEMM (bf16 WMMA).
// One block = one (b, h) row (128 pixels), 256 threads = 8 waves.
// Per tap k (9 total): stage a full 64-channel tile (128 px x 64 bf16 = 16 KB,
// geometry computed once per tap) into a double-buffered LDS tile, then each
// wave runs 2 K-chunks x 4 M-tiles = 8 bf16 WMMAs. One barrier per tap; the
// next tap's gathers are issued before the current tap's WMMAs for overlap.
// LDS addresses XOR-swizzle the 16B group index with (pixel & 7) to spread
// the 128 B pixel rows across all 64 banks.
// ---------------------------------------------------------------------------
__global__ __launch_bounds__(256) void deform_gemm_kernel(
    const float* __restrict__ x,
    const unsigned short* __restrict__ wA,   // packed bf16 A-fragments
    const float* __restrict__ dyw,
    const float* __restrict__ dxw,
    const float* __restrict__ mw,
    const float* __restrict__ bias,
    float* __restrict__ out)
{
    __shared__ __align__(16) unsigned int sm[2][WW * 32];   // 2 x 16 KB

    const int b    = blockIdx.x >> 7;
    const int h    = blockIdx.x & 127;
    const int t    = threadIdx.x;
    const int wave = t >> 5;
    const int lane = t & 31;
    const int p    = t & 127;            // staging pixel
    const int cb   = (t >> 7) * 32;      // staging channel base: 0 / 32

    v8f acc0 = {}, acc1 = {}, acc2 = {}, acc3 = {};
    const float* xb = x + b * CC * HWSZ;

    // ---- stage one tap tile (32 channels of one pixel per thread) ----
    auto stage = [&](int k, int buf) {
        const int ky = k / 3, kx = k % 3;
        const int oidx = (b * KK2 + k) * HWSZ + h * WW + p;
        const float dy = dyw[oidx];
        const float dx = dxw[oidx];
        const float mk = mw[oidx];
        const float ys = dy + (float)(h + ky - 1);
        const float xs = dx + (float)(p + kx - 1);
        const float y0f = floorf(ys), x0f = floorf(xs);
        const float wy1 = ys - y0f, wy0 = 1.0f - wy1;
        const float wx1 = xs - x0f, wx0 = 1.0f - wx1;
        const int y0 = (int)y0f, x0i = (int)x0f;
        const int y1 = y0 + 1,  x1i = x0i + 1;
        float v00 = ((unsigned)y0 < HH && (unsigned)x0i < WW) ? wy0 * wx0 : 0.0f;
        float v01 = ((unsigned)y0 < HH && (unsigned)x1i < WW) ? wy0 * wx1 : 0.0f;
        float v10 = ((unsigned)y1 < HH && (unsigned)x0i < WW) ? wy1 * wx0 : 0.0f;
        float v11 = ((unsigned)y1 < HH && (unsigned)x1i < WW) ? wy1 * wx1 : 0.0f;
        v00 *= mk; v01 *= mk; v10 *= mk; v11 *= mk;
        const int y0c = min(max(y0, 0), HH - 1), y1c = min(max(y1, 0), HH - 1);
        const int x0c = min(max(x0i, 0), WW - 1), x1c = min(max(x1i, 0), WW - 1);
        const int i00 = y0c * WW + x0c, i01 = y0c * WW + x1c;
        const int i10 = y1c * WW + x0c, i11 = y1c * WW + x1c;
        const float* pc = xb + cb * HWSZ;
        unsigned int* smb = &sm[buf][p * 32];
#pragma unroll
        for (int j = 0; j < 16; ++j) {
            const float* pA = pc + (2 * j) * HWSZ;
            const float* pB = pA + HWSZ;
            float fa = pA[i00] * v00 + pA[i01] * v01 + pA[i10] * v10 + pA[i11] * v11;
            float fb = pB[i00] * v00 + pB[i01] * v01 + pB[i10] * v10 + pB[i11] * v11;
            const int w  = (cb >> 1) + j;                       // u32 word 0..31
            const int ws_ = (w & 3) | ((((w >> 2) ^ p) & 7) << 2); // XOR group swizzle
            smb[ws_] = (unsigned int)f2bf(fa) | ((unsigned int)f2bf(fb) << 16);
        }
    };

    // ---- consume one tap tile: 2 chunks x 4 M-tiles = 8 WMMAs ----
    const int pix = wave * 16 + (lane & 15);
    const int hi  = lane >> 4;
    auto consume = [&](int k, int buf) {
        const unsigned int* smb = &sm[buf][pix * 32];
#pragma unroll
        for (int half = 0; half < 2; ++half) {
            const int g0 = half * 4 + hi;
            union { u32x4 q[2]; v16bf v; } bf;
            bf.q[0] = *(const u32x4*)(smb + (((g0     ^ pix) & 7) << 2));
            bf.q[1] = *(const u32x4*)(smb + ((((g0+2) ^ pix) & 7) << 2));
            const int kc = 2 * k + half;
            union { u32x4 q[2]; v16bf v; } af;
            const u32x4* ap;
            ap = (const u32x4*)(wA + ((0 * NCHUNK + kc) * 32 + lane) * 16);
            af.q[0] = ap[0]; af.q[1] = ap[1];
            acc0 = __builtin_amdgcn_wmma_f32_16x16x32_bf16(false, af.v, false, bf.v,
                                                           (short)0, acc0, false, false);
            ap = (const u32x4*)(wA + ((1 * NCHUNK + kc) * 32 + lane) * 16);
            af.q[0] = ap[0]; af.q[1] = ap[1];
            acc1 = __builtin_amdgcn_wmma_f32_16x16x32_bf16(false, af.v, false, bf.v,
                                                           (short)0, acc1, false, false);
            ap = (const u32x4*)(wA + ((2 * NCHUNK + kc) * 32 + lane) * 16);
            af.q[0] = ap[0]; af.q[1] = ap[1];
            acc2 = __builtin_amdgcn_wmma_f32_16x16x32_bf16(false, af.v, false, bf.v,
                                                           (short)0, acc2, false, false);
            ap = (const u32x4*)(wA + ((3 * NCHUNK + kc) * 32 + lane) * 16);
            af.q[0] = ap[0]; af.q[1] = ap[1];
            acc3 = __builtin_amdgcn_wmma_f32_16x16x32_bf16(false, af.v, false, bf.v,
                                                           (short)0, acc3, false, false);
        }
    };

    stage(0, 0);
    __syncthreads();
    for (int k = 0; k < KK2; ++k) {
        if (k < KK2 - 1) stage(k + 1, (k + 1) & 1);   // overlap gathers with WMMA
        consume(k, k & 1);
        __syncthreads();
    }

    // ---- epilogue: D fragment -> out[b, o, h, pix] (+bias) ----
    float* ob = out + (b * OO) * HWSZ + h * WW + pix;
#pragma unroll
    for (int r = 0; r < 8; ++r) {
        int o0 = 0  + r + hi * 8;
        int o1 = 16 + r + hi * 8;
        int o2 = 32 + r + hi * 8;
        int o3 = 48 + r + hi * 8;
        ob[o0 * HWSZ] = acc0[r] + bias[o0];
        ob[o1 * HWSZ] = acc1[r] + bias[o1];
        ob[o2 * HWSZ] = acc2[r] + bias[o2];
        ob[o3 * HWSZ] = acc3[r] + bias[o3];
    }
}

// ---------------------------------------------------------------------------
extern "C" void kernel_launch(void* const* d_in, const int* in_sizes, int n_in,
                              void* d_out, int out_size, void* d_ws, size_t ws_size,
                              hipStream_t stream) {
    const float* x      = (const float*)d_in[0];   // (4, 64, 128, 128)
    const float* w_om   = (const float*)d_in[1];   // (27, 64, 3, 3)
    const float* b_om   = (const float*)d_in[2];   // (27,)
    const float* weight = (const float*)d_in[3];   // (64, 64, 3, 3)
    const float* bias   = (const float*)d_in[4];   // (64,)
    float* out = (float*)d_out;

    char* ws = (char*)d_ws;
    unsigned short* wA  = (unsigned short*)(ws + WS_WA);
    float*          wOm = (float*)(ws + WS_WOM);
    float*          dyw = (float*)(ws + WS_DY);
    float*          dxw = (float*)(ws + WS_DX);
    float*          mw  = (float*)(ws + WS_MASK);

    pack_kernel<<<144, 256, 0, stream>>>(weight, w_om, wA, wOm);
    offset_conv_kernel<<<BB * HH, 256, 0, stream>>>(x, wOm, w_om, b_om, dyw, dxw, mw);
    deform_gemm_kernel<<<BB * HH, 256, 0, stream>>>(x, wA, dyw, dxw, mw, bias, out);
}
